// ATTPredictor_66709432041476
// MI455X (gfx1250) — matrix operand order repaired and checked by reading, not compiled, not run on hardware
//
#include <hip/hip_runtime.h>
#include <hip/hip_bf16.h>

typedef __attribute__((ext_vector_type(16))) _Float16 v16h;
typedef __attribute__((ext_vector_type(8)))  float    v8f;

#define CDIM 256
#define TLEN 128
#define NB   64
#define VOC  106
#define MAXL 25

__device__ __forceinline__ float sigf(float x) { return 1.0f / (1.0f + __expf(-x)); }

// Issue one 32B (2 x b128) async global->LDS copy. INST_OFFSET applies to both
// the global and LDS addresses, so offset:16 places the second chunk correctly.
__device__ __forceinline__ void async_copy32(const float* g, const float* lds_p)
{
    unsigned long long ga = (unsigned long long)g;
    unsigned la = (unsigned)(unsigned long long)lds_p;
    asm volatile("global_load_async_to_lds_b128 %0, %1, off"
                 :: "v"(la), "v"(ga) : "memory");
    asm volatile("global_load_async_to_lds_b128 %0, %1, off offset:16"
                 :: "v"(la), "v"(ga) : "memory");
}
__device__ __forceinline__ void wait_async0()
{
    asm volatile("s_wait_asynccnt 0" ::: "memory");
}

// ---------------------------------------------------------------------------
// Tensor Data Mover: load a 64-row x 32-col f32 tile (row stride = rowStride
// elements) into LDS. The TDM pad feature appends 4 dwords after every 32
// dwords stored, producing exactly the [64][36] padded LDS layout we consume.
// ---------------------------------------------------------------------------
#if defined(__has_builtin)
#if __has_builtin(__builtin_amdgcn_tensor_load_to_lds)
#define HAVE_TDM 1
#endif
#endif

#ifdef HAVE_TDM
typedef __attribute__((ext_vector_type(4))) unsigned int u32x4;
typedef __attribute__((ext_vector_type(8))) int          i32x8;
typedef __attribute__((ext_vector_type(4))) int          i32x4;

__device__ __forceinline__ void tdm_load_tile_64x32(const float* gtile, const float* ldst,
                                                    int rowStride)
{
    unsigned long long ga = (unsigned long long)gtile;
    unsigned la = (unsigned)(unsigned long long)ldst;
    const int tensorRows = 1 << 20;   // always in-bounds for our tiles

    u32x4 g0;
    g0[0] = 1u;                                 // count=1 valid descriptor, user mode
    g0[1] = la;                                 // lds_addr (bytes)
    g0[2] = (unsigned)ga;                       // global_addr[31:0]
    g0[3] = (unsigned)(ga >> 32) | (2u << 30);  // global_addr[56:32] | type=2 (image)

    i32x8 g1;
    // data_size=4B (2), pad_enable=1, pad_interval=32 dwords (4), pad_amount=4 dwords (3)
    g1[0] = (2 << 16) | (1 << 20) | (4 << 22) | (3 << 25);
    g1[1] = (rowStride & 0xFFFF) << 16;                            // tensor_dim0 lo16
    g1[2] = ((unsigned)rowStride >> 16) | ((tensorRows & 0xFFFF) << 16); // dim0 hi / dim1 lo
    g1[3] = ((unsigned)tensorRows >> 16) | (32 << 16);             // dim1 hi | tile_dim0=32
    g1[4] = 64;                                                    // tile_dim1=64, tile_dim2=0
    g1[5] = rowStride;                                             // tensor_dim0_stride lo32
    g1[6] = 0;
    g1[7] = 0;

    i32x4 z4 = {0, 0, 0, 0};
#if __clang_major__ >= 23
    i32x8 z8 = {};
    __builtin_amdgcn_tensor_load_to_lds(g0, g1, z4, z4, z8, 0);
#else
    __builtin_amdgcn_tensor_load_to_lds(g0, g1, z4, z4, 0);
#endif
}
#endif // HAVE_TDM

// ---------------------------------------------------------------------------
// Generic WMMA GEMM: C[M,N] = A[M,K] @ W[N,K]^T + bias0 + bias1
// BM=BN=64, BK=32, 256 threads (8 waves), 2 subtiles/wave.
// Tiles staged f32 into LDS via TDM descriptors (one issuing wave, TENSORcnt),
// or per-lane async global->LDS loads as fallback; double buffered either way.
// Fragments convert f32->f16 at read time, feeding v_wmma_f32_16x16x32_f16.
// M,N multiples of 64; K multiple of 32 (true for all call sites).
// ---------------------------------------------------------------------------
__global__ __launch_bounds__(256)
void gemm_wmma_kernel(const float* __restrict__ A, int lda,
                      const float* __restrict__ W, int ldw,
                      float* __restrict__ C, int ldc,
                      int M, int N, int K,
                      const float* __restrict__ bias0,
                      const float* __restrict__ bias1)
{
    __shared__ float Asf[2][64][36];   // [buf][m][k]
    __shared__ float Bsf[2][64][36];   // [buf][n][k]
    const int tid  = threadIdx.x;
    const int lane = tid & 31;
    const int wid  = tid >> 5;
    const int m0 = blockIdx.y * 64;
    const int n0 = blockIdx.x * 64;
    const int l15  = lane & 15;
    const int kb   = (lane < 16) ? 0 : 8;   // A frag K base
    const int koff = (lane < 16) ? 0 : 16;  // unused; B frag also uses kb
    const int lrow = tid >> 2;              // 0..63
    const int kseg = (tid & 3) * 8;         // 0,8,16,24

    v8f acc0 = {}, acc1 = {};
    const int nIter = K / 32;

#ifdef HAVE_TDM
    // prologue: one wave issues both tile-0 descriptors (TDM ignores EXEC,
    // so only wave 0 may reach the instruction).
    if (tid < 32) {
        tdm_load_tile_64x32(A + (size_t)m0 * lda, &Asf[0][0][0], lda);
        tdm_load_tile_64x32(W + (size_t)n0 * ldw, &Bsf[0][0][0], ldw);
    }
#else
    async_copy32(A + (size_t)(m0 + lrow) * lda + kseg, &Asf[0][lrow][kseg]);
    async_copy32(W + (size_t)(n0 + lrow) * ldw + kseg, &Bsf[0][lrow][kseg]);
#endif

    for (int it = 0; it < nIter; ++it) {
        const int buf = it & 1;
#ifdef HAVE_TDM
        if (tid < 32) __builtin_amdgcn_s_wait_tensorcnt(0);
        __syncthreads();
        if (it + 1 < nIter && tid < 32) {   // software pipeline: next tile
            const int k0 = (it + 1) * 32;
            tdm_load_tile_64x32(A + (size_t)m0 * lda + k0, &Asf[buf ^ 1][0][0], lda);
            tdm_load_tile_64x32(W + (size_t)n0 * ldw + k0, &Bsf[buf ^ 1][0][0], ldw);
        }
#else
        wait_async0();
        __syncthreads();
        if (it + 1 < nIter) {   // software pipeline: prefetch next tile
            const int k0 = (it + 1) * 32;
            async_copy32(A + (size_t)(m0 + lrow) * lda + k0 + kseg, &Asf[buf ^ 1][lrow][kseg]);
            async_copy32(W + (size_t)(n0 + lrow) * ldw + k0 + kseg, &Bsf[buf ^ 1][lrow][kseg]);
        }
#endif
#pragma unroll
        for (int half = 0; half < 2; ++half) {
            const int st = wid + half * 8;       // 0..15
            const int mi = st >> 2, ni = st & 3;
            const int row = mi * 16 + l15;
            const int col = ni * 16 + l15;
            v16h a, b;
#pragma unroll
            for (int e = 0; e < 8; ++e) {
                a[e]     = (_Float16)Asf[buf][row][kb + e];
                a[8 + e] = (_Float16)Asf[buf][row][kb + 16 + e];
            }
#pragma unroll
            for (int e = 0; e < 8; ++e) {
                b[e]     = (_Float16)Bsf[buf][col][kb + e];
                b[8 + e] = (_Float16)Bsf[buf][col][kb + 16 + e];
            }
            if (half == 0)
                acc0 = __builtin_amdgcn_wmma_f32_16x16x32_f16(false, a, false, b, (short)0, acc0, false, false);
            else
                acc1 = __builtin_amdgcn_wmma_f32_16x16x32_f16(false, a, false, b, (short)0, acc1, false, false);
        }
        __syncthreads();
    }
#pragma unroll
    for (int half = 0; half < 2; ++half) {
        const int st = wid + half * 8;
        const int mi = st >> 2, ni = st & 3;
        const v8f acc = half ? acc1 : acc0;
        const int col = n0 + ni * 16 + l15;
        float bsum = 0.0f;                 // hoisted: one conditional load per column
        if (bias0) bsum += bias0[col];
        if (bias1) bsum += bias1[col];
#pragma unroll
        for (int r = 0; r < 8; ++r) {
            int row = m0 + mi * 16 + r + ((lane >> 4) << 3);
            C[(size_t)row * ldc + col] = acc[r] + bsum;
        }
    }
}

// ---------------------------------------------------------------------------
// Implicit-GEMM conv: OUT[n,oc,oh,ow] = sum_{ic,kh,kw} X[n,ic,2*oh+kh-1,ow+kw-1]*W[oc,ic,kh,kw]
// K = 256*9 = 2304. grid = (spatialTiles, 4 ocTiles, 64 batch).
// Weight tile staged via async global->LDS (f32); im2col tile gathered
// manually (zero padding) while the async load is in flight.
// ---------------------------------------------------------------------------
__global__ __launch_bounds__(256)
void conv_wmma_kernel(const float* __restrict__ X, const float* __restrict__ Wt,
                      float* __restrict__ Out, int Hin, int Hout)
{
    __shared__ float    Asf[64][36];     // [oc][k]
    __shared__ _Float16 Bs[64][40];      // [s][k]
    const int tid  = threadIdx.x;
    const int lane = tid & 31;
    const int wid  = tid >> 5;
    const int n  = blockIdx.z;
    const int m0 = blockIdx.y * 64;         // oc base
    const int s0 = blockIdx.x * 64;         // spatial base (64 positions, same oh)
    const int oh = s0 >> 7;
    const int owb = s0 & 127;
    const int l15  = lane & 15;
    const int kb   = (lane < 16) ? 0 : 8;
    const int koff = (lane < 16) ? 0 : 16;
    const int Nsp = Hout * 128;

    v8f acc0 = {}, acc1 = {};
    const int lrow = tid >> 2, kseg = (tid & 3) * 8;
    const int lkk = tid >> 3;               // 0..31
    const int lsj = (tid & 7) * 8;          // 0..56

    for (int k0 = 0; k0 < 2304; k0 += 32) {
        // async-stage weight tile (overlaps with im2col gather below)
        async_copy32(Wt + (size_t)(m0 + lrow) * 2304 + k0 + kseg, &Asf[lrow][kseg]);
        // im2col gather for B tile: Bs[s][k]
        {
            int k  = k0 + lkk;
            int ic = k / 9;
            int r9 = k - ic * 9;
            int kh = r9 / 3;
            int kw = r9 - kh * 3;
            int ih = oh * 2 + kh - 1;
            const float* xrow = X + (((size_t)n * CDIM + ic) * Hin + ih) * 128;
            bool hok = (ih >= 0) && (ih < Hin);
#pragma unroll
            for (int e = 0; e < 8; ++e) {
                int iw = owb + lsj + e + kw - 1;
                float v = (hok && iw >= 0 && iw < 128) ? xrow[iw] : 0.0f;
                Bs[lsj + e][lkk] = (_Float16)v;
            }
        }
        wait_async0();
        __syncthreads();
#pragma unroll
        for (int half = 0; half < 2; ++half) {
            const int st = wid + half * 8;
            const int mi = st >> 2, ni = st & 3;
            const int row = mi * 16 + l15;
            const int col = ni * 16 + l15;
            v16h a, b;
#pragma unroll
            for (int e = 0; e < 8; ++e) {
                a[e]     = (_Float16)Asf[row][kb + e];
                a[8 + e] = (_Float16)Asf[row][kb + 16 + e];
            }
#pragma unroll
            for (int e = 0; e < 8; ++e) {
                b[e]     = Bs[col][kb + e];
                b[8 + e] = Bs[col][kb + 16 + e];
            }
            if (half == 0)
                acc0 = __builtin_amdgcn_wmma_f32_16x16x32_f16(false, a, false, b, (short)0, acc0, false, false);
            else
                acc1 = __builtin_amdgcn_wmma_f32_16x16x32_f16(false, a, false, b, (short)0, acc1, false, false);
        }
        __syncthreads();
    }
#pragma unroll
    for (int half = 0; half < 2; ++half) {
        const int st = wid + half * 8;
        const int mi = st >> 2, ni = st & 3;
        const v8f acc = half ? acc1 : acc0;
#pragma unroll
        for (int r = 0; r < 8; ++r) {
            int oc = m0 + mi * 16 + r + ((lane >> 4) << 3);
            int s  = s0 + ni * 16 + l15;
            Out[((size_t)n * CDIM + oc) * Nsp + s] = acc[r];
        }
    }
}

// ---------------------------------------------------------------------------
// GroupNorm stats: one block per (n, g); reduce 8 channels x HW.
// ---------------------------------------------------------------------------
__global__ __launch_bounds__(256)
void gn_stats_kernel(const float* __restrict__ x, float* __restrict__ stats, int HW)
{
    const int nb = blockIdx.x >> 5;
    const int g  = blockIdx.x & 31;
    const float* base = x + ((size_t)nb * CDIM + g * 8) * HW;
    const int cnt = 8 * HW;
    float s = 0.0f, sq = 0.0f;
    for (int i = threadIdx.x; i < cnt; i += 256) {
        float v = base[i];
        s += v; sq += v * v;
    }
    __shared__ float ssum[256], ssq[256];
    ssum[threadIdx.x] = s; ssq[threadIdx.x] = sq;
    __syncthreads();
    for (int off = 128; off > 0; off >>= 1) {
        if (threadIdx.x < off) {
            ssum[threadIdx.x] += ssum[threadIdx.x + off];
            ssq[threadIdx.x]  += ssq[threadIdx.x + off];
        }
        __syncthreads();
    }
    if (threadIdx.x == 0) {
        float m   = ssum[0] / (float)cnt;
        float var = ssq[0] / (float)cnt - m * m;
        stats[2 * blockIdx.x]     = m;
        stats[2 * blockIdx.x + 1] = rsqrtf(var + 1e-5f);
    }
}

__global__ __launch_bounds__(256)
void gn_apply_relu_kernel(float* __restrict__ x, const float* __restrict__ stats,
                          const float* __restrict__ gamma, const float* __restrict__ beta,
                          int HW, int total)
{
    int i = blockIdx.x * 256 + threadIdx.x;
    if (i >= total) return;
    int ch = (i / HW) & 255;
    int nb = i / (CDIM * HW);
    int g  = ch >> 3;
    float m = stats[(nb * 32 + g) * 2];
    float r = stats[(nb * 32 + g) * 2 + 1];
    float v = (x[i] - m) * r * gamma[ch] + beta[ch];
    x[i] = v > 0.0f ? v : 0.0f;
}

// mean over H (4) + transpose (N,C,4,128) -> (T=128, N, C)
__global__ __launch_bounds__(256)
void mean_t_kernel(const float* __restrict__ y, float* __restrict__ xt)
{
    int i = blockIdx.x * 256 + threadIdx.x;   // 0..2097151
    int ch = i & 255;
    int nb = (i >> 8) & 63;
    int t  = i >> 14;
    const float* p = y + (((size_t)nb * CDIM + ch) * 4) * 128 + t;
    xt[i] = 0.25f * (p[0] + p[128] + p[256] + p[384]);
}

// ---------------------------------------------------------------------------
// Persistent BiLSTM: grid=2 (dir), 1024 threads = 32 waves.
// Per step: gates[64,1024] = xpre[t] + h @ Whh^T (WMMA), then cell update.
// Writes hidden states into rec[t][n][dir*256 + j].
// ---------------------------------------------------------------------------
__global__ __launch_bounds__(1024)
void lstm_kernel(const float* __restrict__ xpre_f, const float* __restrict__ xpre_b,
                 const float* __restrict__ whh_f,  const float* __restrict__ whh_b,
                 float* __restrict__ rec, float* __restrict__ scr)
{
    const int dir = blockIdx.x;
    const float* xpre = dir ? xpre_b : xpre_f;
    const float* whh  = dir ? whh_b  : whh_f;
    float* h     = scr + dir * 16384;
    float* c     = scr + 32768 + dir * 16384;
    float* gates = scr + 65536 + dir * 65536;
    const int tid  = threadIdx.x;
    const int lane = tid & 31;
    const int wid  = tid >> 5;
    const int l15  = lane & 15;
    const int kb   = (lane < 16) ? 0 : 8;
    const int koff = (lane < 16) ? 0 : 16;
    const int rofs = ((lane >> 4) << 3);

    for (int i = tid; i < 16384; i += 1024) { h[i] = 0.0f; c[i] = 0.0f; }
    __syncthreads();

    for (int tt = 0; tt < TLEN; ++tt) {
        const int t = dir ? (TLEN - 1 - tt) : tt;
        const float* xp = xpre + (size_t)t * NB * 1024;
        for (int tile = wid; tile < 256; tile += 32) {
            const int mi = tile >> 6;      // 0..3
            const int nj = tile & 63;      // 0..63
            const int rbase = mi * 16, cbase = nj * 16;
            v8f acc;
#pragma unroll
            for (int r = 0; r < 8; ++r)
                acc[r] = xp[(size_t)(rbase + r + rofs) * 1024 + cbase + l15];
#pragma unroll
            for (int k0 = 0; k0 < CDIM; k0 += 32) {
                const float* hrow = h + (rbase + l15) * CDIM + k0;
                const float* wrow = whh + (size_t)(cbase + l15) * CDIM + k0;
                v16h a, b;
#pragma unroll
                for (int e = 0; e < 8; ++e) {
                    a[e]     = (_Float16)hrow[kb + e];
                    a[8 + e] = (_Float16)hrow[kb + 16 + e];
                }
#pragma unroll
                for (int e = 0; e < 8; ++e) {
                    b[e]     = (_Float16)wrow[kb + e];
                    b[8 + e] = (_Float16)wrow[kb + 16 + e];
                }
                acc = __builtin_amdgcn_wmma_f32_16x16x32_f16(false, a, false, b, (short)0, acc, false, false);
            }
#pragma unroll
            for (int r = 0; r < 8; ++r)
                gates[(rbase + r + rofs) * 1024 + cbase + l15] = acc[r];
        }
        __syncthreads();
        for (int i = tid; i < 16384; i += 1024) {
            const int row = i >> 8, j = i & 255;
            float gi = gates[row * 1024 + j];
            float gf = gates[row * 1024 + 256 + j];
            float gg = gates[row * 1024 + 512 + j];
            float go = gates[row * 1024 + 768 + j];
            float cc = sigf(gf) * c[i] + sigf(gi) * tanhf(gg);
            float hh = sigf(go) * tanhf(cc);
            c[i] = cc; h[i] = hh;
            rec[((size_t)t * NB + row) * 512 + dir * CDIM + j] = hh;
        }
        __syncthreads();
    }
}

// ---------------------------------------------------------------------------
// Persistent attention-GRU decoder: 1 WG, 1024 threads, 25 sequential steps.
// ---------------------------------------------------------------------------
__global__ __launch_bounds__(1024)
void decoder_kernel(const float* __restrict__ enc, const float* __restrict__ att_emb,
                    const float* __restrict__ comb_w, const float* __restrict__ comb_b,
                    const float* __restrict__ gru_wih, const float* __restrict__ gru_whh,
                    const float* __restrict__ gru_bih, const float* __restrict__ gru_bhh,
                    const float* __restrict__ out_w, const float* __restrict__ out_b,
                    const float* __restrict__ vat_w, const float* __restrict__ vat_b,
                    float* __restrict__ out, float* __restrict__ scr, int* __restrict__ inp)
{
    float* h   = scr;
    float* ctx = scr + 16384;
    float* o   = scr + 32768;
    float* gi  = scr + 49152;
    float* gh  = scr + 98304;
    float* aw  = scr + 147456;
    float* lg  = scr + 155648;
    const int tid = threadIdx.x;
    for (int i = tid; i < 16384; i += 1024) h[i] = 0.0f;
    if (tid < NB) inp[tid] = 0;
    __syncthreads();
    const float vb = vat_b[0];

    for (int step = 0; step < MAXL; ++step) {
        // attention scores: aw[t][n] = sum_c tanh(h + enc) * vat_w
        for (int p = tid; p < TLEN * NB; p += 1024) {
            const int t = p >> 6, nb = p & 63;
            const float* ep = enc + ((size_t)t * NB + nb) * CDIM;
            const float* hp = h + nb * CDIM;
            float s = vb;
            for (int cc = 0; cc < CDIM; ++cc) s += tanhf(hp[cc] + ep[cc]) * vat_w[cc];
            aw[t * NB + nb] = s;
        }
        __syncthreads();
        if (tid < NB) {  // softmax over time
            float mx = -1e30f;
            for (int t = 0; t < TLEN; ++t) mx = fmaxf(mx, aw[t * NB + tid]);
            float ss = 0.0f;
            for (int t = 0; t < TLEN; ++t) { float e = __expf(aw[t * NB + tid] - mx); aw[t * NB + tid] = e; ss += e; }
            float inv = 1.0f / ss;
            for (int t = 0; t < TLEN; ++t) aw[t * NB + tid] *= inv;
        }
        __syncthreads();
        for (int i = tid; i < 16384; i += 1024) {   // context
            const int nb = i >> 8, cc = i & 255;
            float s = 0.0f;
            for (int t = 0; t < TLEN; ++t) s += aw[t * NB + nb] * enc[((size_t)t * NB + nb) * CDIM + cc];
            ctx[i] = s;
        }
        __syncthreads();
        for (int i = tid; i < 16384; i += 1024) {   // combine + relu
            const int nb = i >> 8, cc = i & 255;
            const float* e  = att_emb + inp[nb] * CDIM;
            const float* w0 = comb_w + (size_t)cc * 512;
            float s = comb_b[cc];
            for (int j = 0; j < CDIM; ++j) s += e[j] * w0[j];
            for (int j = 0; j < CDIM; ++j) s += ctx[nb * CDIM + j] * w0[CDIM + j];
            o[i] = s > 0.0f ? s : 0.0f;
        }
        __syncthreads();
        for (int i = tid; i < NB * 768; i += 1024) { // GRU gates
            const int nb = i / 768, g = i % 768;
            const float* wi = gru_wih + (size_t)g * CDIM;
            const float* wh = gru_whh + (size_t)g * CDIM;
            const float* op = o + nb * CDIM;
            const float* hp = h + nb * CDIM;
            float si = gru_bih[g], sh = gru_bhh[g];
            for (int j = 0; j < CDIM; ++j) { si += op[j] * wi[j]; sh += hp[j] * wh[j]; }
            gi[i] = si; gh[i] = sh;
        }
        __syncthreads();
        for (int i = tid; i < 16384; i += 1024) {   // h update
            const int nb = i >> 8, j = i & 255;
            float r  = sigf(gi[nb * 768 + j] + gh[nb * 768 + j]);
            float z  = sigf(gi[nb * 768 + 256 + j] + gh[nb * 768 + 256 + j]);
            float nn = tanhf(gi[nb * 768 + 512 + j] + r * gh[nb * 768 + 512 + j]);
            h[i] = (1.0f - z) * nn + z * h[i];
        }
        __syncthreads();
        for (int i = tid; i < NB * VOC; i += 1024) { // logits
            const int nb = i / VOC, v = i % VOC;
            const float* wv = out_w + (size_t)v * CDIM;
            const float* hp = h + nb * CDIM;
            float s = out_b[v];
            for (int j = 0; j < CDIM; ++j) s += hp[j] * wv[j];
            lg[i] = s;
        }
        __syncthreads();
        if (tid < NB) {  // log_softmax + argmax + emit
            float mx = -1e30f; int am = 0;
            for (int v = 0; v < VOC; ++v) { float x = lg[tid * VOC + v]; if (x > mx) { mx = x; am = v; } }
            float ss = 0.0f;
            for (int v = 0; v < VOC; ++v) ss += __expf(lg[tid * VOC + v] - mx);
            float lse = mx + __logf(ss);
            float* op = out + ((size_t)tid * MAXL + step) * VOC;
            for (int v = 0; v < VOC; ++v) op[v] = lg[tid * VOC + v] - lse;
            inp[tid] = am;
        }
        __syncthreads();
    }
}

// ---------------------------------------------------------------------------
extern "C" void kernel_launch(void* const* d_in, const int* in_sizes, int n_in,
                              void* d_out, int out_size, void* d_ws, size_t ws_size,
                              hipStream_t stream) {
    const float* rois    = (const float*)d_in[0];
    const float* conv1_w = (const float*)d_in[1];
    const float* gn1_g   = (const float*)d_in[2];
    const float* gn1_b   = (const float*)d_in[3];
    const float* conv2_w = (const float*)d_in[4];
    const float* gn2_g   = (const float*)d_in[5];
    const float* gn2_b   = (const float*)d_in[6];
    const float* wih_f   = (const float*)d_in[7];
    const float* whh_f   = (const float*)d_in[8];
    const float* bih_f   = (const float*)d_in[9];
    const float* bhh_f   = (const float*)d_in[10];
    const float* wih_b   = (const float*)d_in[11];
    const float* whh_b   = (const float*)d_in[12];
    const float* bih_b   = (const float*)d_in[13];
    const float* bhh_b   = (const float*)d_in[14];
    const float* emb_w   = (const float*)d_in[15];
    const float* emb_b   = (const float*)d_in[16];
    const float* att_emb = (const float*)d_in[17];
    const float* comb_w  = (const float*)d_in[18];
    const float* comb_b  = (const float*)d_in[19];
    const float* gru_wih = (const float*)d_in[20];
    const float* gru_whh = (const float*)d_in[21];
    const float* gru_bih = (const float*)d_in[22];
    const float* gru_bhh = (const float*)d_in[23];
    const float* out_w   = (const float*)d_in[24];
    const float* out_b   = (const float*)d_in[25];
    const float* vat_w   = (const float*)d_in[26];
    const float* vat_b   = (const float*)d_in[27];

    float* ws = (float*)d_ws;
    float* conv1b   = ws;                          // 16,777,216 f (reused for xpre)
    float* conv2b   = ws + 16777216;               //  8,388,608 f
    float* xt       = ws + 25165824;               //  2,097,152 f
    float* rec      = ws + 27262976;               //  4,194,304 f
    float* enc      = ws + 31457280;               //  2,097,152 f
    float* stats    = ws + 33554432;               //      4,096 f
    float* lstm_scr = ws + 33558528;               //    196,608 f
    float* dec_scr  = ws + 33755136;               //    162,432 f
    int*   inp      = (int*)(ws + 33917568);
    float* xpre_f   = conv1b;                      // aliases conv1 (dead by then)
    float* xpre_b   = conv1b + 8388608;

    dim3 b256(256), b1024(1024);

    // conv1 (64,256,16,128) -> (64,256,8,128)
    conv_wmma_kernel<<<dim3(16, 4, 64), b256, 0, stream>>>(rois, conv1_w, conv1b, 16, 8);
    gn_stats_kernel<<<2048, b256, 0, stream>>>(conv1b, stats, 1024);
    gn_apply_relu_kernel<<<65536, b256, 0, stream>>>(conv1b, stats, gn1_g, gn1_b, 1024, 16777216);
    // conv2 -> (64,256,4,128)
    conv_wmma_kernel<<<dim3(8, 4, 64), b256, 0, stream>>>(conv1b, conv2_w, conv2b, 8, 4);
    gn_stats_kernel<<<2048, b256, 0, stream>>>(conv2b, stats, 512);
    gn_apply_relu_kernel<<<32768, b256, 0, stream>>>(conv2b, stats, gn2_g, gn2_b, 512, 8388608);
    // mean over H + transpose -> (T,N,C)
    mean_t_kernel<<<8192, b256, 0, stream>>>(conv2b, xt);
    // xpre = x @ Wih^T + (bih + bhh), both directions (8192x1024x256)
    gemm_wmma_kernel<<<dim3(16, 128), b256, 0, stream>>>(xt, 256, wih_f, 256, xpre_f, 1024,
                                                         8192, 1024, 256, bih_f, bhh_f);
    gemm_wmma_kernel<<<dim3(16, 128), b256, 0, stream>>>(xt, 256, wih_b, 256, xpre_b, 1024,
                                                         8192, 1024, 256, bih_b, bhh_b);
    // persistent BiLSTM recurrence
    lstm_kernel<<<2, b1024, 0, stream>>>(xpre_f, xpre_b, whh_f, whh_b, rec, lstm_scr);
    // enc = rec @ emb_w^T + emb_b (8192x256x512)
    gemm_wmma_kernel<<<dim3(4, 128), b256, 0, stream>>>(rec, 512, emb_w, 512, enc, 256,
                                                        8192, 256, 512, emb_b, nullptr);
    // attention-GRU decoder (25 sequential steps, greedy feedback)
    decoder_kernel<<<1, b1024, 0, stream>>>(enc, att_emb, comb_w, comb_b, gru_wih, gru_whh,
                                            gru_bih, gru_bhh, out_w, out_b, vat_w, vat_b,
                                            (float*)d_out, dec_scr, inp);
}
